// SSRvNNEncoder_24713241821881
// MI455X (gfx1250) — matrix operand order repaired and checked by reading, not compiled, not run on hardware
//
#include <hip/hip_runtime.h>
#include <hip/hip_bf16.h>

// ---------------- problem constants (from reference) ----------------
#define DIM   256          // D == H
#define BB    16           // batch
#define TT    8            // num_subtrees
#define LL    512          // subtree_len
#define EE    2048         // edges per graph
#define GG    (BB*TT)      // 128 graphs
#define GL    (GG*LL)      // 65536 node rows
#define NTPB  256

typedef __attribute__((ext_vector_type(16))) __bf16 v16bf;
typedef __attribute__((ext_vector_type(8)))  float  v8f;
typedef __attribute__((ext_vector_type(4)))  unsigned int u32x4;
typedef __attribute__((ext_vector_type(8)))  int          i32x8;
typedef __attribute__((ext_vector_type(4)))  int          i32x4;

__device__ __forceinline__ unsigned short f2bf(float f) {
    unsigned u = __float_as_uint(f);
    u += 0x7FFFu + ((u >> 16) & 1u);       // round-to-nearest-even
    return (unsigned short)(u >> 16);
}

__device__ __forceinline__ void atomicMaxF(float* addr, float val) {
    unsigned* ua = (unsigned*)addr;
    unsigned old = *ua;
    while (__uint_as_float(old) < val) {
        unsigned assumed = old;
        old = atomicCAS(ua, assumed, __float_as_uint(val));
        if (old == assumed) break;
    }
}

// -------------------------------------------------------------------
// K1: embedding gather  xg[g,l,:] = emb_table[token_ids[t*L+l, b]]
// -------------------------------------------------------------------
__global__ void k_gather(const int* __restrict__ token_ids,
                         const float* __restrict__ emb,
                         float* __restrict__ xg32,
                         unsigned short* __restrict__ xgbf)
{
    const int r = blockIdx.x;          // node row in [0, G*L)
    const int d = threadIdx.x;         // [0, 256)
    const int g = r >> 9;              // / L
    const int l = r & (LL - 1);
    const int b = g >> 3;              // / T
    const int t = g & (TT - 1);
    const int tok = token_ids[(t * LL + l) * BB + b];
    const float v = emb[(size_t)tok * DIM + d];
    xg32[(size_t)r * DIM + d] = v;
    xgbf[(size_t)r * DIM + d] = f2bf(v);
}

// -------------------------------------------------------------------
// K1b: transpose-convert weight:  Wt[n][k] (bf16) = W[k][n] (f32)
// -------------------------------------------------------------------
__global__ void k_convWt(const float* __restrict__ W,
                         unsigned short* __restrict__ Wt)
{
    const int n = blockIdx.x;          // 0..255
    const int k = threadIdx.x;         // 0..255
    Wt[n * DIM + k] = f2bf(W[k * DIM + n]);
}

// -------------------------------------------------------------------
// K2: C[M,256] = A[M,256] @ W[256,256] + bias  via v_wmma_f32_16x16x32_bf16
//     - Wt (bf16, transposed [n][k]) tile staged into LDS by the
//       Tensor Data Mover (tensor_load_to_lds + s_wait_tensorcnt).
//     - block = 8 waves -> 64x32 output tile.
// -------------------------------------------------------------------
__global__ void __launch_bounds__(NTPB)
k_gemm_bias(const unsigned short* __restrict__ A,    // bf16 bits, M x 256
            const unsigned short* __restrict__ Wt,   // bf16 bits, 256 x 256 [n][k]
            const float* __restrict__ bias,          // 256
            float* __restrict__ C)                   // M x 256
{
    __shared__ unsigned short bsT[32 * 256];         // Wt rows nBase..nBase+31
    const int tid   = threadIdx.x;
    const int lane  = tid & 31;
    const int wave  = tid >> 5;
    const int wm    = wave & 3;                      // M subtile 0..3
    const int wn    = wave >> 2;                     // N subtile 0..1
    const int mBase = blockIdx.x * 64;
    const int nBase = blockIdx.y * 32;

    // ---- TDM: DMA Wt[nBase:nBase+32, 0:256] (bf16) into LDS ----
    if (wave == 0) {
        const unsigned long long ga =
            (unsigned long long)(uintptr_t)(Wt + (size_t)nBase * DIM);
        const unsigned ldsOff = (unsigned)(uintptr_t)&bsT[0];  // low 32b = LDS offset

        u32x4 g0;
        g0.x = 1u;                                    // count=1, user mode
        g0.y = ldsOff;                                // lds_addr (bytes)
        g0.z = (unsigned)(ga & 0xFFFFFFFFu);          // global_addr[31:0]
        g0.w = (unsigned)((ga >> 32) & 0x1FFFFFFu)    // global_addr[56:32]
             | (2u << 30);                            // type = 2 (image)

        i32x8 g1;
        g1[0] = 0x00010000;          // workgroup_mask=0, data_size=1 (2B)
        g1[1] = (int)(0x0100u << 16);// tensor_dim0[15:0]=256 in bits 63:48
        g1[2] = (int)(0x0100u << 16);// dim0 hi=0 | tensor_dim1[15:0]=256
        g1[3] = (int)(256u << 16);   // dim1 hi=0 | tile_dim0=256
        g1[4] = 32;                  // tile_dim1=32, tile_dim2=0
        g1[5] = 256;                 // tensor_dim0_stride = 256 (low 32)
        g1[6] = 0;                   // stride hi / dim1_stride lo
        g1[7] = 0;
        i32x4 gz4 = {0, 0, 0, 0};
        i32x8 gz8 = {0, 0, 0, 0, 0, 0, 0, 0};
        __builtin_amdgcn_tensor_load_to_lds(g0, g1, gz4, gz4, gz8, 0);
    }
    __builtin_amdgcn_s_wait_tensorcnt(0);
    __syncthreads();

    const int mRow  = mBase + wm * 16 + (lane & 15);
    const int khalf = lane >> 4;                     // K-half select
    const int ncol  = wn * 16 + (lane & 15);         // column in 32-wide tile

    const uint4* arow = (const uint4*)(A + (size_t)mRow * DIM);     // 8 bf16/chunk
    const uint4* brow = (const uint4*)(bsT + ncol * DIM);

    v8f acc = {};
    #pragma unroll
    for (int kk = 0; kk < 8; ++kk) {
        if (kk < 7)  // prefetch next A K-slice (global_prefetch_b8)
            __builtin_prefetch(A + (size_t)mRow * DIM + kk * 32 + 32, 0, 1);

        union { v16bf v; uint4 q[2]; } ua, ub;
        // A 16x32 bf16 fragment: lanes0-15 K = {0..7,16..23}+k0 ; lanes16-31 +8
        ua.q[0] = arow[kk * 4 + khalf];
        ua.q[1] = arow[kk * 4 + 2 + khalf];
        // B 32x16 bf16 fragment: lanes0-15 K = {0..7,16..23}+k0 ; lanes16-31 +8
        ub.q[0] = brow[kk * 4 + khalf];
        ub.q[1] = brow[kk * 4 + 2 + khalf];

        acc = __builtin_amdgcn_wmma_f32_16x16x32_bf16(
                  false, ua.v, false, ub.v, (short)0, acc, false, false);
    }

    #pragma unroll
    for (int r = 0; r < 8; ++r) {
        const int row = mBase + wm * 16 + ((lane < 16) ? r : (r + 8));
        const int col = nBase + ncol;
        C[(size_t)row * DIM + col] = acc[r] + bias[col];
    }
}

// -------------------------------------------------------------------
// K3: per-graph edge weights:  w=||xs-xd||, beta=mean(w), ew=exp(-w^2/2b^2)
// -------------------------------------------------------------------
__global__ void k_edgew(const float* __restrict__ xg,
                        const int* __restrict__ edge_index,
                        float* __restrict__ ew)
{
    const int g = blockIdx.x;
    __shared__ float wsh[EE];
    __shared__ float red[NTPB];
    const int* src = edge_index + (size_t)g * 2 * EE;
    const int* dst = src + EE;

    float local = 0.f;
    for (int e = threadIdx.x; e < EE; e += NTPB) {
        const float* ps = xg + ((size_t)g * LL + src[e]) * DIM;
        const float* pd = xg + ((size_t)g * LL + dst[e]) * DIM;
        float acc = 0.f;
        for (int k = 0; k < DIM; ++k) { const float df = ps[k] - pd[k]; acc += df * df; }
        const float w = sqrtf(acc + 1e-12f);
        wsh[e] = w;
        local += w;
    }
    red[threadIdx.x] = local;
    __syncthreads();
    for (int s = NTPB / 2; s > 0; s >>= 1) {
        if (threadIdx.x < s) red[threadIdx.x] += red[threadIdx.x + s];
        __syncthreads();
    }
    const float beta = red[0] * (1.0f / (float)EE);
    const float inv  = 1.0f / (2.0f * beta * beta);
    for (int e = threadIdx.x; e < EE; e += NTPB) {
        const float w = wsh[e];
        ew[(size_t)g * EE + e] = expf(-w * w * inv);
    }
}

// -------------------------------------------------------------------
// K4: score[e] = leaky_relu(xl[src] + xr[dst] + ew*We, 0.2) . att
// -------------------------------------------------------------------
__global__ void k_score(const float* __restrict__ xl,
                        const float* __restrict__ xr,
                        const float* __restrict__ ew,
                        const int* __restrict__ edge_index,
                        const float* __restrict__ We,
                        const float* __restrict__ att,
                        float* __restrict__ score)
{
    const int idx  = blockIdx.x * 8 + (threadIdx.x >> 5);   // edge in [0, G*E)
    const int lane = threadIdx.x & 31;
    const int g = idx / EE, e = idx % EE;
    const int s = edge_index[(size_t)g * 2 * EE + e];
    const int d = edge_index[(size_t)g * 2 * EE + EE + e];
    const float w = ew[(size_t)g * EE + e];
    const float* pl = xl + ((size_t)g * LL + s) * DIM;
    const float* pr = xr + ((size_t)g * LL + d) * DIM;

    float part = 0.f;
    for (int k = lane; k < DIM; k += 32) {
        float f = pl[k] + pr[k] + w * We[k];
        f = (f > 0.f) ? f : 0.2f * f;
        part += f * att[k];
    }
    for (int off = 16; off; off >>= 1) part += __shfl_xor(part, off, 32);
    if (lane == 0) score[(size_t)g * EE + e] = part;
}

// -------------------------------------------------------------------
// K5: init segment buffers + agg
// -------------------------------------------------------------------
__global__ void k_seg_init(float* __restrict__ seg_max,
                           float* __restrict__ seg_den,
                           float* __restrict__ agg)
{
    const size_t total = (size_t)GL * DIM;
    for (size_t i = blockIdx.x * (size_t)NTPB + threadIdx.x; i < total;
         i += (size_t)gridDim.x * NTPB) {
        agg[i] = 0.f;
        if (i < GL) { seg_max[i] = -INFINITY; seg_den[i] = 0.f; }
    }
}

// K6a: segment max (by dst node within graph)
__global__ void k_seg_max(const float* __restrict__ score,
                          const int* __restrict__ edge_index,
                          float* __restrict__ seg_max)
{
    const int i = blockIdx.x * NTPB + threadIdx.x;     // edge
    const int g = i / EE, e = i % EE;
    const int d = edge_index[(size_t)g * 2 * EE + EE + e];
    atomicMaxF(&seg_max[g * LL + d], score[i]);
}

// K6b: ex = exp(s - m); score <- ex ; den += ex
__global__ void k_seg_den(float* __restrict__ score,
                          const int* __restrict__ edge_index,
                          const float* __restrict__ seg_max,
                          float* __restrict__ seg_den)
{
    const int i = blockIdx.x * NTPB + threadIdx.x;
    const int g = i / EE, e = i % EE;
    const int d = edge_index[(size_t)g * 2 * EE + EE + e];
    const float m  = seg_max[g * LL + d];
    const float ex = expf(score[i] - m);
    score[i] = ex;
    atomicAdd(&seg_den[g * LL + d], ex);
}

// K7: scatter  agg[g,dst,:] += (ex/den) * xl[g,src,:]
__global__ void k_scatter(const float* __restrict__ score,   // holds ex
                          const float* __restrict__ seg_den,
                          const float* __restrict__ xl,
                          const int* __restrict__ edge_index,
                          float* __restrict__ agg)
{
    const size_t i    = (size_t)blockIdx.x * NTPB + threadIdx.x;
    const int    edge = (int)(i >> 8);        // / DIM
    const int    d    = (int)(i & (DIM - 1));
    const int    g    = edge / EE, e = edge % EE;
    const int    sn   = edge_index[(size_t)g * 2 * EE + e];
    const int    dn   = edge_index[(size_t)g * 2 * EE + EE + e];
    const float alpha = score[edge] / (seg_den[g * LL + dn] + 1e-16f);
    const float val   = alpha * xl[((size_t)g * LL + sn) * DIM + d];
    atomicAdd(&agg[((size_t)g * LL + dn) * DIM + d], val);
}

// K8: pooled[g,d] = max_l agg[g,l,d] + gat_bias[d]
__global__ void k_pool(const float* __restrict__ agg,
                       const float* __restrict__ gat_bias,
                       float* __restrict__ pooled)
{
    const int g = blockIdx.x, d = threadIdx.x;
    float m = -INFINITY;
    for (int l = 0; l < LL; ++l)
        m = fmaxf(m, agg[((size_t)g * LL + l) * DIM + d]);
    pooled[(size_t)g * DIM + d] = m + gat_bias[d];
}

// -------------------------------------------------------------------
// K9: GRU, one block per direction; hidden state in LDS.
// -------------------------------------------------------------------
__global__ void k_gru(const float* __restrict__ pooled,      // (B,T,D) == (g,D)
                      const float* __restrict__ Wih_f, const float* __restrict__ Whh_f,
                      const float* __restrict__ bih_f, const float* __restrict__ bhh_f,
                      const float* __restrict__ Wih_b, const float* __restrict__ Whh_b,
                      const float* __restrict__ bih_b, const float* __restrict__ bhh_b,
                      float* __restrict__ ys_f, float* __restrict__ ys_b,
                      float* __restrict__ d_out)              // hidden at T*B*H
{
    const int dir = blockIdx.x;
    const float* Wih = dir ? Wih_b : Wih_f;
    const float* Whh = dir ? Whh_b : Whh_f;
    const float* bih = dir ? bih_b : bih_f;
    const float* bhh = dir ? bhh_b : bhh_f;
    float* ys = dir ? ys_b : ys_f;

    __shared__ float h [BB * DIM];
    __shared__ float hn[BB * DIM];
    for (int i = threadIdx.x; i < BB * DIM; i += NTPB) h[i] = 0.f;
    __syncthreads();

    const int j = threadIdx.x;                         // hidden index
    for (int step = 0; step < TT; ++step) {
        const int t = dir ? (TT - 1 - step) : step;
        for (int b = 0; b < BB; ++b) {
            const float* x  = pooled + ((size_t)(b * TT + t)) * DIM;  // seq[t,b,:]
            const float* hb = h + b * DIM;
            float gir = bih[j],       ghr = bhh[j];
            float giz = bih[DIM + j], ghz = bhh[DIM + j];
            float gin = bih[2*DIM+j], ghn = bhh[2*DIM+j];
            for (int k = 0; k < DIM; ++k) {
                const float xv = x[k], hv = hb[k];
                gir += xv * Wih[(0 * DIM + j) * DIM + k];
                giz += xv * Wih[(1 * DIM + j) * DIM + k];
                gin += xv * Wih[(2 * DIM + j) * DIM + k];
                ghr += hv * Whh[(0 * DIM + j) * DIM + k];
                ghz += hv * Whh[(1 * DIM + j) * DIM + k];
                ghn += hv * Whh[(2 * DIM + j) * DIM + k];
            }
            const float r = 1.f / (1.f + expf(-(gir + ghr)));
            const float z = 1.f / (1.f + expf(-(giz + ghz)));
            const float n = tanhf(gin + r * ghn);
            const float hnew = (1.f - z) * n + z * hb[j];
            hn[b * DIM + j] = hnew;
            ys[(size_t)t * BB * DIM + b * DIM + j] = hnew;
        }
        __syncthreads();
        for (int i = threadIdx.x; i < BB * DIM; i += NTPB) h[i] = hn[i];
        __syncthreads();
    }
    for (int b = 0; b < BB; ++b)
        d_out[(size_t)TT * BB * DIM + (size_t)dir * BB * DIM + b * DIM + j] =
            h[b * DIM + j];
}

// K10: outputs = ys_f + ys_b
__global__ void k_sum_out(const float* __restrict__ ys_f,
                          const float* __restrict__ ys_b,
                          float* __restrict__ d_out)
{
    const int i = blockIdx.x * NTPB + threadIdx.x;
    d_out[i] = ys_f[i] + ys_b[i];
}

// -------------------------------------------------------------------
extern "C" void kernel_launch(void* const* d_in, const int* in_sizes, int n_in,
                              void* d_out, int out_size, void* d_ws, size_t ws_size,
                              hipStream_t stream)
{
    const int*   token_ids = (const int*)  d_in[0];
    const int*   edge_idx  = (const int*)  d_in[1];
    const float* emb       = (const float*)d_in[2];
    const float* Wl        = (const float*)d_in[3];
    const float* bl        = (const float*)d_in[4];
    const float* Wr        = (const float*)d_in[5];
    const float* br        = (const float*)d_in[6];
    const float* We        = (const float*)d_in[7];
    const float* att       = (const float*)d_in[8];
    const float* gat_bias  = (const float*)d_in[9];
    const float* Wih_f     = (const float*)d_in[10];
    const float* Whh_f     = (const float*)d_in[11];
    const float* bih_f     = (const float*)d_in[12];
    const float* bhh_f     = (const float*)d_in[13];
    const float* Wih_b     = (const float*)d_in[14];
    const float* Whh_b     = (const float*)d_in[15];
    const float* bih_b     = (const float*)d_in[16];
    const float* bhh_b     = (const float*)d_in[17];
    float* out = (float*)d_out;

    // workspace carve-up
    char* p = (char*)d_ws;
    auto carve = [&](size_t bytes) -> char* {
        char* r = p; p += (bytes + 255) & ~(size_t)255; return r;
    };
    float*          xg32   = (float*)         carve((size_t)GL * DIM * 4);
    unsigned short* xgbf   = (unsigned short*)carve((size_t)GL * DIM * 2);
    float*          xl     = (float*)         carve((size_t)GL * DIM * 4);
    float*          xr     = (float*)         carve((size_t)GL * DIM * 4);
    float*          agg    = (float*)         carve((size_t)GL * DIM * 4);
    float*          ew     = (float*)         carve((size_t)GG * EE * 4);
    float*          score  = (float*)         carve((size_t)GG * EE * 4);
    float*          segm   = (float*)         carve((size_t)GL * 4);
    float*          segd   = (float*)         carve((size_t)GL * 4);
    float*          pooled = (float*)         carve((size_t)GG * DIM * 4);
    float*          ysf    = (float*)         carve((size_t)TT * BB * DIM * 4);
    float*          ysb    = (float*)         carve((size_t)TT * BB * DIM * 4);
    unsigned short* WlT    = (unsigned short*)carve((size_t)DIM * DIM * 2);
    unsigned short* WrT    = (unsigned short*)carve((size_t)DIM * DIM * 2);

    // 1. gather embeddings (+bf16 copy); weight transpose-convert
    k_gather<<<GL, NTPB, 0, stream>>>(token_ids, emb, xg32, xgbf);
    k_convWt<<<DIM, DIM, 0, stream>>>(Wl, WlT);
    k_convWt<<<DIM, DIM, 0, stream>>>(Wr, WrT);

    // 2. WMMA GEMMs: xl, xr (TDM-staged weight tiles)
    dim3 gg(GL / 64, DIM / 32);
    k_gemm_bias<<<gg, NTPB, 0, stream>>>(xgbf, WlT, bl, xl);
    k_gemm_bias<<<gg, NTPB, 0, stream>>>(xgbf, WrT, br, xr);

    // 3. edge weights
    k_edgew<<<GG, NTPB, 0, stream>>>(xg32, edge_idx, ew);

    // 4. attention scores
    k_score<<<(GG * EE) / 8, NTPB, 0, stream>>>(xl, xr, ew, edge_idx, We, att, score);

    // 5-7. segment softmax + scatter aggregation
    k_seg_init<<<4096, NTPB, 0, stream>>>(segm, segd, agg);
    k_seg_max<<<(GG * EE) / NTPB, NTPB, 0, stream>>>(score, edge_idx, segm);
    k_seg_den<<<(GG * EE) / NTPB, NTPB, 0, stream>>>(score, edge_idx, segm, segd);
    k_scatter<<<((size_t)GG * EE * DIM) / NTPB, NTPB, 0, stream>>>(
        score, segd, xl, edge_idx, agg);

    // 8. max-pool over L (+gat_bias)
    k_pool<<<GG, NTPB, 0, stream>>>(agg, gat_bias, pooled);

    // 9. bidirectional GRU
    k_gru<<<2, NTPB, 0, stream>>>(pooled,
                                  Wih_f, Whh_f, bih_f, bhh_f,
                                  Wih_b, Whh_b, bih_b, bhh_b,
                                  ysf, ysb, out);

    // 10. outputs = ys_f + ys_b
    k_sum_out<<<(TT * BB * DIM) / NTPB, NTPB, 0, stream>>>(ysf, ysb, out);
}
